// PershomLearnedFilt_82222853914922
// MI455X (gfx1250) — matrix operand order repaired and checked by reading, not compiled, not run on hardware
//
#include <hip/hip_runtime.h>
#include <hip/hip_bf16.h>

// Problem constants (match reference setup_inputs)
#define NNODE 300000
#define NEDGE 2400000
#define BATCH 512
#define KC    100
#define P0N   260000
#define P0EN  512
#define P1EN  40000
#define BN_EPS 1e-5f

typedef __attribute__((ext_vector_type(16))) __bf16          v16bf;
typedef __attribute__((ext_vector_type(8)))  float           v8f;
typedef __attribute__((ext_vector_type(8)))  unsigned short  v8us;
typedef __attribute__((ext_vector_type(16))) unsigned short  v16us;

// ---------- bf16 helpers ----------
__device__ __forceinline__ unsigned short bfbits(float f) {
  unsigned int u = __builtin_bit_cast(unsigned int, f);
  u += 0x7fffu + ((u >> 16) & 1u);      // RNE
  return (unsigned short)(u >> 16);
}
__device__ __forceinline__ float bf2f(unsigned short s) {
  unsigned int u = ((unsigned int)s) << 16;
  return __builtin_bit_cast(float, u);
}

// ---------- generic zero ----------
__global__ void k_zero(float* p, int n) {
  int t = blockIdx.x * blockDim.x + threadIdx.x;
  if (t < n) p[t] = 0.f;
}

// ---------- weight repack: fp32 [K,64] row-major -> bf16 WMMA B-fragments ----------
// Per (kt, nt) tile of 32(K)x16(N): element index = lane*16 + e,
// B layout (16-bit, 32x16): lane L holds col N = L%16, k = (L/16)*8 + (e&7) + (e/8)*16.
__global__ void k_packW(const float* __restrict__ W, unsigned short* __restrict__ Wp, int K) {
  int t = blockIdx.x * blockDim.x + threadIdx.x;
  int total = K * 64;
  if (t >= total) return;
  int e    = t & 15;
  int lane = (t >> 4) & 31;
  int nt   = (t >> 9) & 3;
  int kt   = t >> 11;
  int n = nt * 16 + (lane & 15);
  int k = kt * 32 + (lane >> 4) * 8 + (e & 7) + (e >> 3) * 16;
  Wp[t] = bfbits(W[k * 64 + n]);
}

// ---------- embedding gather: x = concat(emb_deg[deg], emb_lab[lab]) -> bf16 slab ----------
__global__ void k_embed(const int* __restrict__ deg, const int* __restrict__ lab,
                        const float* __restrict__ ed, const float* __restrict__ el,
                        unsigned short* __restrict__ xcbf, int N) {
  int t = blockIdx.x * blockDim.x + threadIdx.x;
  if (t >= N * 128) return;
  int i = t >> 7, j = t & 127;
  float v = (j < 64) ? ed[deg[i] * 64 + j] : el[lab[i] * 64 + (j - 64)];
  xcbf[(size_t)i * 320 + j] = bfbits(v);
}

// ---------- s = self copy (bf16 slab slice -> fp32 accumulation buffer) ----------
__global__ void k_init_s(const unsigned short* __restrict__ xcbf, int srcOff,
                         float* __restrict__ s, int Din, int N) {
  int t = blockIdx.x * blockDim.x + threadIdx.x;
  if (t >= N * Din) return;
  int i = t / Din, j = t - i * Din;
  s[t] = bf2f(xcbf[(size_t)i * 320 + srcOff + j]);
}

// ---------- edge scatter-add: s[dst] += z[src]  (hardware f32 atomics) ----------
__global__ void k_agg(const int* __restrict__ ei, int E,
                      const unsigned short* __restrict__ xcbf, int srcOff,
                      float* __restrict__ s, int Din) {
  int per = Din >> 2;                 // 4 floats per thread
  int t = blockIdx.x * blockDim.x + threadIdx.x;
  if (t >= E * per) return;
  int e = t / per;
  int q = (t - e * per) << 2;
  int sn = ei[e];
  int dn = ei[E + e];
  const unsigned short* sp = xcbf + (size_t)sn * 320 + srcOff + q;
  float* dp = s + (size_t)dn * Din + q;
  unsafeAtomicAdd(dp + 0, bf2f(sp[0]));
  unsafeAtomicAdd(dp + 1, bf2f(sp[1]));
  unsafeAtomicAdd(dp + 2, bf2f(sp[2]));
  unsafeAtomicAdd(dp + 3, bf2f(sp[3]));
}

// ---------- LDS fragment load (two 16B halves; robust to 16B LDS alignment) ----------
__device__ __forceinline__ v16bf ld_frag_lds(const unsigned short* p) {
  v8us lo = *(const v8us*)(p);
  v8us hi = *(const v8us*)(p + 8);
  v16us u;
#pragma unroll
  for (int e = 0; e < 8; ++e) { u[e] = lo[e]; u[e + 8] = hi[e]; }
  return __builtin_bit_cast(v16bf, u);
}

// ---------- WMMA GEMM: C[N,64] = A[N,K] @ W[K,64] + bias ----------
// One wave computes a 16(M) x 64(N) tile with 4 accumulators, K-loop step 32.
// Weight panel (K*64 bf16, <=40KB) staged to LDS via gfx1250 async copies.
// A fragment (16-bit A 16x32): lane L holds row L%16; grp = L/16;
//   elems 0..7 -> k = grp*8 + e, elems 8..15 -> k = 16 + grp*8 + (e-8).
template <bool ABF16>
__global__ void k_gemm64(const void* __restrict__ Av,
                         const unsigned short* __restrict__ Wp,
                         const float* __restrict__ bias,
                         float* __restrict__ C, int N, int K) {
  extern __shared__ unsigned short smem[];
  // ---- async global -> LDS staging of the packed weight panel ----
  // chunks = K*64/8 is a multiple of blockDim (256) for K in {64,128,320},
  // so every async issue happens with full EXEC and no divergence.
  {
    const int chunks = (K * 64) >> 3;                  // 16-byte chunks
    for (int c = threadIdx.x; c < chunks; c += blockDim.x) {
      unsigned lds = (unsigned)(size_t)(&smem[c << 3]);     // low 32 bits = LDS offset
      unsigned long long ga = (unsigned long long)(Wp + ((size_t)c << 3));
      asm volatile("global_load_async_to_lds_b128 %0, %1, off"
                   :: "v"(lds), "v"(ga) : "memory");
    }
    asm volatile("s_wait_asynccnt 0" ::: "memory");
    __syncthreads();
  }

  const int wave = (int)((blockIdx.x * blockDim.x + threadIdx.x) >> 5);
  const int lane = (int)(threadIdx.x & 31);
  const int m0 = wave << 4;
  if (m0 >= N) return;                 // wave-uniform: EXEC stays all-ones
  const int grp  = lane >> 4;
  const int lrow = lane & 15;
  int row = m0 + lrow;
  if (row >= N) row = N - 1;           // branch-free clamp (rows always mapped)
  v8f acc[4] = {};
  const int nkt = K >> 5;
  for (int kt = 0; kt < nkt; ++kt) {
    v16bf a;
    if constexpr (ABF16) {
      const unsigned short* p = (const unsigned short*)Av + (size_t)row * K + (kt << 5) + (grp << 3);
      v8us lo = *(const v8us*)(p);         // k = grp*8 + 0..7
      v8us hi = *(const v8us*)(p + 16);    // k = 16 + grp*8 + 0..7
      v16us u;
#pragma unroll
      for (int e = 0; e < 8; ++e) { u[e] = lo[e]; u[e + 8] = hi[e]; }
      a = __builtin_bit_cast(v16bf, u);
      if (kt + 1 < nkt) __builtin_prefetch(p + 32, 0, 1);
    } else {
      const float* p = (const float*)Av + (size_t)row * K + (kt << 5) + (grp << 3);
      v8f f0 = *(const v8f*)(p);           // k = grp*8 + 0..7
      v8f f1 = *(const v8f*)(p + 16);      // k = 16 + grp*8 + 0..7
#pragma unroll
      for (int e = 0; e < 8; ++e) { a[e] = (__bf16)f0[e]; a[e + 8] = (__bf16)f1[e]; }
      if (kt + 1 < nkt) __builtin_prefetch(p + 32, 0, 1);
    }
    const unsigned short* wpt = smem + ((size_t)kt << 11) + lane * 16;
    v16bf b0 = ld_frag_lds(wpt);
    v16bf b1 = ld_frag_lds(wpt + 512);
    v16bf b2 = ld_frag_lds(wpt + 1024);
    v16bf b3 = ld_frag_lds(wpt + 1536);
    acc[0] = __builtin_amdgcn_wmma_f32_16x16x32_bf16(false, a, false, b0, (short)0, acc[0], false, false);
    acc[1] = __builtin_amdgcn_wmma_f32_16x16x32_bf16(false, a, false, b1, (short)0, acc[1], false, false);
    acc[2] = __builtin_amdgcn_wmma_f32_16x16x32_bf16(false, a, false, b2, (short)0, acc[2], false, false);
    acc[3] = __builtin_amdgcn_wmma_f32_16x16x32_bf16(false, a, false, b3, (short)0, acc[3], false, false);
  }
  // C/D layout: lane L, VGPR v -> row = m0 + grp*8 + v, col = nt*16 + L%16
#pragma unroll
  for (int nt = 0; nt < 4; ++nt) {
    int c = nt * 16 + lrow;
    float bc = bias[c];
#pragma unroll
    for (int v = 0; v < 8; ++v) {
      int r = m0 + grp * 8 + v;
      if (r < N) C[(size_t)r * 64 + c] = acc[nt][v] + bc;
    }
  }
}

// ---------- per-column sum/sumsq over [N,64] (LDS partial + 1 atomic per col/block) ----------
__global__ void k_colstats(const float* __restrict__ h, int N,
                           float* __restrict__ sums, float* __restrict__ sqs) {
  __shared__ float ls[256], lq[256];
  int col = threadIdx.x & 63;
  int r0 = blockIdx.x * 4 + (threadIdx.x >> 6);
  float s = 0.f, q = 0.f;
  for (int r = r0; r < N; r += gridDim.x * 4) {
    float v = h[(size_t)r * 64 + col];
    s += v; q += v * v;
  }
  ls[threadIdx.x] = s; lq[threadIdx.x] = q;
  __syncthreads();
  if (threadIdx.x < 64) {
    s = ls[threadIdx.x] + ls[threadIdx.x + 64] + ls[threadIdx.x + 128] + ls[threadIdx.x + 192];
    q = lq[threadIdx.x] + lq[threadIdx.x + 64] + lq[threadIdx.x + 128] + lq[threadIdx.x + 192];
    unsafeAtomicAdd(&sums[col], s);
    unsafeAtomicAdd(&sqs[col], q);
  }
}

// ---------- BN (training fwd, biased var) + LeakyReLU; write bf16 slab and/or fp32 ----------
__global__ void k_bn_act(const float* __restrict__ h, const float* __restrict__ sums,
                         const float* __restrict__ sqs, float invN,
                         const float* __restrict__ g, const float* __restrict__ be,
                         unsigned short* out_bf, int dstOff, float* out_f, int N) {
  int t = blockIdx.x * blockDim.x + threadIdx.x;
  if (t >= N * 64) return;
  int i = t >> 6, j = t & 63;
  float mean = sums[j] * invN;
  float var  = sqs[j] * invN - mean * mean;
  float v = (h[t] - mean) * rsqrtf(var + BN_EPS) * g[j] + be[j];
  v = (v >= 0.f) ? v : 0.01f * v;
  if (out_bf) out_bf[(size_t)i * 320 + dstOff + j] = bfbits(v);
  if (out_f)  out_f[t] = v;
}

// ---------- node filtration: nf = sigmoid(hf @ fW2 + fb2) ----------
__global__ void k_filt(const float* __restrict__ hf, const float* __restrict__ fW2,
                       const float* __restrict__ fb2, float* __restrict__ nf, int N) {
  int i = blockIdx.x * blockDim.x + threadIdx.x;
  if (i >= N) return;
  float a = fb2[0];
  const float* hp = hf + (size_t)i * 64;
#pragma unroll 8
  for (int j = 0; j < 64; ++j) a += hp[j] * fW2[j];
  nf[i] = 1.f / (1.f + __expf(-a));
}

// ---------- rational-hat persistence layer, scatter into feat[B,300] ----------
__global__ void k_slayer(const float* __restrict__ nf, const int* __restrict__ idx, int P, int d,
                         const int* __restrict__ sample, const float* __restrict__ centers,
                         const float* __restrict__ radius, float* __restrict__ feat, int off) {
  __shared__ float cs[2 * KC];
  for (int t = threadIdx.x; t < KC * d; t += blockDim.x) cs[t] = centers[t];
  __syncthreads();
  float r = fabsf(radius[0]);
  int p = blockIdx.x * blockDim.x + threadIdx.x;
  if (p >= P) return;
  float x0 = nf[idx[p * d]];
  float x1 = (d == 2) ? nf[idx[p * d + 1]] : 0.f;
  float* fp = feat + (size_t)sample[p] * 300 + off;
  for (int k = 0; k < KC; ++k) {
    float dist = fabsf(x0 - cs[k * d]);
    if (d == 2) dist += fabsf(x1 - cs[k * d + 1]);
    float f = 1.f / (1.f + dist) - 1.f / (1.f + r + dist);
    unsafeAtomicAdd(fp + k, f);
  }
}

// ---------- classifier ----------
__global__ void k_cgemm1(const float* __restrict__ feat, const float* __restrict__ cW1,
                         const float* __restrict__ cb1, float* __restrict__ cfeat) {
  int t = blockIdx.x * blockDim.x + threadIdx.x;
  if (t >= BATCH * 100) return;
  int b = t / 100, c = t - b * 100;
  float acc = cb1[c];
  const float* fp = feat + (size_t)b * 300;
  for (int k = 0; k < 300; ++k) acc += fp[k] * cW1[k * 100 + c];
  cfeat[t] = acc;
}
__global__ void k_cstats(const float* __restrict__ cfeat, float* __restrict__ mean,
                         float* __restrict__ var) {
  int c = threadIdx.x;
  if (c >= 100) return;
  float s = 0.f, q = 0.f;
  for (int r = 0; r < BATCH; ++r) { float v = cfeat[r * 100 + c]; s += v; q += v * v; }
  float m = s / (float)BATCH;
  mean[c] = m;
  var[c] = q / (float)BATCH - m * m;
}
__global__ void k_cact(float* __restrict__ cfeat, const float* __restrict__ mean,
                       const float* __restrict__ var, const float* __restrict__ cg,
                       const float* __restrict__ cbe) {
  int t = blockIdx.x * blockDim.x + threadIdx.x;
  if (t >= BATCH * 100) return;
  int c = t % 100;
  float v = (cfeat[t] - mean[c]) * rsqrtf(var[c] + BN_EPS) * cg[c] + cbe[c];
  cfeat[t] = (v >= 0.f) ? v : 0.01f * v;
}
__global__ void k_cout(const float* __restrict__ cfeat, const float* __restrict__ cW2,
                       const float* __restrict__ cb2, float* __restrict__ out) {
  int t = blockIdx.x * blockDim.x + threadIdx.x;
  if (t >= BATCH * 10) return;
  int b = t / 10, o = t - b * 10;
  float acc = cb2[o];
  const float* fp = cfeat + (size_t)b * 100;
  for (int j = 0; j < 100; ++j) acc += fp[j] * cW2[j * 10 + o];
  out[t] = acc;
}

static inline int cdiv(long a, long b) { return (int)((a + b - 1) / b); }

extern "C" void kernel_launch(void* const* d_in, const int* in_sizes, int n_in,
                              void* d_out, int out_size, void* d_ws, size_t ws_size,
                              hipStream_t stream) {
  const int*   node_deg   = (const int*)d_in[0];
  const int*   node_lab   = (const int*)d_in[1];
  const int*   edge_index = (const int*)d_in[2];
  const int*   h0_idx     = (const int*)d_in[3];
  const int*   h0_sample  = (const int*)d_in[4];
  const int*   h0e_idx    = (const int*)d_in[5];
  const int*   h0e_sample = (const int*)d_in[6];
  const int*   h1e_idx    = (const int*)d_in[7];
  const int*   h1e_sample = (const int*)d_in[8];
  const float* emb_deg    = (const float*)d_in[9];
  const float* emb_lab    = (const float*)d_in[10];
  const float* W0  = (const float*)d_in[11];
  const float* b0  = (const float*)d_in[12];
  const float* g0  = (const float*)d_in[13];
  const float* be0 = (const float*)d_in[14];
  const float* W1  = (const float*)d_in[15];
  const float* b1  = (const float*)d_in[16];
  const float* g1  = (const float*)d_in[17];
  const float* be1 = (const float*)d_in[18];
  const float* W2  = (const float*)d_in[19];
  const float* b2  = (const float*)d_in[20];
  const float* g2  = (const float*)d_in[21];
  const float* be2 = (const float*)d_in[22];
  const float* fW1 = (const float*)d_in[23];
  const float* fb1 = (const float*)d_in[24];
  const float* fg  = (const float*)d_in[25];
  const float* fbe = (const float*)d_in[26];
  const float* fW2 = (const float*)d_in[27];
  const float* fb2 = (const float*)d_in[28];
  const float* c0  = (const float*)d_in[29];
  const float* r0  = (const float*)d_in[30];
  const float* c1  = (const float*)d_in[31];
  const float* r1  = (const float*)d_in[32];
  const float* c2  = (const float*)d_in[33];
  const float* r2  = (const float*)d_in[34];
  const float* cW1 = (const float*)d_in[35];
  const float* cb1 = (const float*)d_in[36];
  const float* cg  = (const float*)d_in[37];
  const float* cbe = (const float*)d_in[38];
  const float* cW2 = (const float*)d_in[39];
  const float* cb2 = (const float*)d_in[40];
  float* out = (float*)d_out;

  // -------- workspace bump allocator (256B aligned) --------
  char* ws = (char*)d_ws;
  size_t off = 0;
  auto alloc = [&](size_t bytes) -> void* {
    void* p = ws + off;
    off = (off + bytes + 255) & ~(size_t)255;
    return p;
  };
  unsigned short* xcbf = (unsigned short*)alloc((size_t)NNODE * 320 * 2); // bf16 concat [x|z1|z2|z3]
  float* s  = (float*)alloc((size_t)NNODE * 128 * 4);   // self+agg accumulator; reused as hf
  float* h  = (float*)alloc((size_t)NNODE * 64 * 4);    // GEMM output
  float* nf = (float*)alloc((size_t)NNODE * 4);         // node filtration values
  unsigned short* wp0 = (unsigned short*)alloc(128 * 64 * 2);
  unsigned short* wp1 = (unsigned short*)alloc(64 * 64 * 2);
  unsigned short* wp2 = (unsigned short*)alloc(64 * 64 * 2);
  unsigned short* wpf = (unsigned short*)alloc(320 * 64 * 2);
  float* stats = (float*)alloc(128 * 4);                // sums[64] | sqs[64]
  float* feat  = (float*)alloc((size_t)BATCH * 300 * 4);
  float* cfeat = (float*)alloc((size_t)BATCH * 100 * 4);
  float* cstat = (float*)alloc(200 * 4);                // mean[100] | var[100]

  const int TB = 256;
  const float invN = 1.f / (float)NNODE;

  // -------- pack weights into WMMA B-fragment layout --------
  k_packW<<<cdiv(128 * 64, TB), TB, 0, stream>>>(W0, wp0, 128);
  k_packW<<<cdiv(64 * 64, TB),  TB, 0, stream>>>(W1, wp1, 64);
  k_packW<<<cdiv(64 * 64, TB),  TB, 0, stream>>>(W2, wp2, 64);
  k_packW<<<cdiv(320 * 64, TB), TB, 0, stream>>>(fW1, wpf, 320);

  // -------- node features --------
  k_embed<<<cdiv((long)NNODE * 128, TB), TB, 0, stream>>>(node_deg, node_lab, emb_deg, emb_lab, xcbf, NNODE);

  const int gemmBlocks = cdiv(NNODE / 16, 8); // 8 waves/block, 16 rows/wave

  // -------- GIN layers --------
  struct { int srcOff, Din, dstOff; const unsigned short* wp;
           const float *b, *g, *be; } L[3] = {
    {0,   128, 128, wp0, b0, g0, be0},
    {128, 64,  192, wp1, b1, g1, be1},
    {192, 64,  256, wp2, b2, g2, be2},
  };
  for (int l = 0; l < 3; ++l) {
    int Din = L[l].Din;
    k_init_s<<<cdiv((long)NNODE * Din, TB), TB, 0, stream>>>(xcbf, L[l].srcOff, s, Din, NNODE);
    k_agg<<<cdiv((long)NEDGE * (Din / 4), TB), TB, 0, stream>>>(edge_index, NEDGE, xcbf, L[l].srcOff, s, Din);
    k_gemm64<false><<<gemmBlocks, TB, Din * 64 * 2, stream>>>(s, L[l].wp, L[l].b, h, NNODE, Din);
    k_zero<<<1, 128, 0, stream>>>(stats, 128);
    k_colstats<<<512, TB, 0, stream>>>(h, NNODE, stats, stats + 64);
    k_bn_act<<<cdiv((long)NNODE * 64, TB), TB, 0, stream>>>(h, stats, stats + 64, invN,
                                                            L[l].g, L[l].be,
                                                            xcbf, L[l].dstOff, nullptr, NNODE);
  }

  // -------- filtration head: xc[N,320] @ fW1 -> BN -> lrelu -> @fW2 -> sigmoid --------
  k_gemm64<true><<<gemmBlocks, TB, 320 * 64 * 2, stream>>>(xcbf, wpf, fb1, h, NNODE, 320);
  k_zero<<<1, 128, 0, stream>>>(stats, 128);
  k_colstats<<<512, TB, 0, stream>>>(h, NNODE, stats, stats + 64);
  k_bn_act<<<cdiv((long)NNODE * 64, TB), TB, 0, stream>>>(h, stats, stats + 64, invN,
                                                          fg, fbe, nullptr, 0, s, NNODE);
  k_filt<<<cdiv(NNODE, TB), TB, 0, stream>>>(s, fW2, fb2, nf, NNODE);

  // -------- persistence layers --------
  k_zero<<<cdiv(BATCH * 300, TB), TB, 0, stream>>>(feat, BATCH * 300);
  k_slayer<<<cdiv(P0N, TB),  TB, 0, stream>>>(nf, h0_idx,  P0N,  2, h0_sample,  c0, r0, feat, 0);
  k_slayer<<<cdiv(P0EN, TB), TB, 0, stream>>>(nf, h0e_idx, P0EN, 1, h0e_sample, c1, r1, feat, 100);
  k_slayer<<<cdiv(P1EN, TB), TB, 0, stream>>>(nf, h1e_idx, P1EN, 1, h1e_sample, c2, r2, feat, 200);

  // -------- classifier --------
  k_cgemm1<<<cdiv(BATCH * 100, TB), TB, 0, stream>>>(feat, cW1, cb1, cfeat);
  k_cstats<<<1, 128, 0, stream>>>(cfeat, cstat, cstat + 100);
  k_cact<<<cdiv(BATCH * 100, TB), TB, 0, stream>>>(cfeat, cstat, cstat + 100, cg, cbe);
  k_cout<<<cdiv(BATCH * 10, TB), TB, 0, stream>>>(cfeat, cW2, cb2, out);

  (void)in_sizes; (void)n_in; (void)out_size; (void)ws_size;
}
